// MultiHeadSelfAttention_47485158425128
// MI455X (gfx1250) — compile-verified, matrix operand order checked by reading, not run on hardware
//
#include <hip/hip_runtime.h>
#include <hip/hip_bf16.h>

// ---------------------------------------------------------------------------
// Multi-head self-attention for MI455X (gfx1250), wave32 + WMMA f16 path,
// with async global->LDS staging (CDNA5 ASYNCcnt path) in the GEMMs.
// B=2, S=2048, H=16, D=64, HD=1024.
// ---------------------------------------------------------------------------

typedef __attribute__((ext_vector_type(16))) _Float16 v16h;
typedef __attribute__((ext_vector_type(8)))  _Float16 v8h;
typedef __attribute__((ext_vector_type(8)))  float    v8f;

#define S_LEN 2048
#define HEADS 16
#define DEPTH 64
#define HDIM  1024
#define MROWS 4096   // B * S

#define BLK_M 128
#define BLK_N 64
#define BLK_K 32

__device__ __forceinline__ v8f wmma_f16(v16h a, v16h b, v8f c) {
    return __builtin_amdgcn_wmma_f32_16x16x32_f16(
        /*neg_a=*/false, a, /*neg_b=*/false, b,
        /*c_mod=*/(short)0, c, /*reuse_a=*/false, /*reuse_b=*/false);
}

// Load a 16-f16 WMMA fragment for one lane.
// 16-bit A/B VGPR layout: lanes 0-15 hold K = [0..7] and [16..23],
// lanes 16-31 hold K = [8..15] and [24..31]. Caller passes
// p = rowBase + kBase + kOff  where kOff = (lane>>4)*8.
__device__ __forceinline__ v16h load_frag(const _Float16* __restrict__ p) {
    v8h lo = *(const v8h*)(p);
    v8h hi = *(const v8h*)(p + 16);
    v16h r;
#pragma unroll
    for (int i = 0; i < 8; ++i) { r[i] = lo[i]; r[i + 8] = hi[i]; }
    return r;
}

// ---- CDNA5 async global->LDS copy (ASYNCcnt), per-lane 16B ----------------
// VDST = VGPR holding LDS byte address, VADDR = 64-bit global address.
// Flat pointers to LDS have the group offset in the low 32 bits.
__device__ __forceinline__ uint32_t lds_addr(const void* p) {
    return (uint32_t)(uintptr_t)p;
}
__device__ __forceinline__ void async_copy_b128(uint32_t lds_off, const void* gaddr) {
    asm volatile("global_load_async_to_lds_b128 %0, %1, off"
                 :: "v"(lds_off), "v"(gaddr)
                 : "memory");
}
__device__ __forceinline__ void wait_async0() {
    asm volatile("s_wait_asynccnt 0x0" ::: "memory");
}

// ---------------------------------------------------------------------------
// f32 -> f16 conversion
// ---------------------------------------------------------------------------
__global__ __launch_bounds__(256) void cvt_f32_f16(const float* __restrict__ in,
                                                   _Float16* __restrict__ out,
                                                   int n) {
    int i = blockIdx.x * 256 + threadIdx.x;
    if (i < n) out[i] = (_Float16)in[i];
}

// ---------------------------------------------------------------------------
// GEMM: y[m,n] = sum_k A[m,k] * W[n,k] + bias[n]
//   A: [4096,1024] f16 row-major, W: [1024,1024] f16 row-major.
// mode 0: out f16, layout [B,H,S,D]      (Q, K)
// mode 1: out f16, layout [B,H,D,S]      (V transposed)
// mode 2: out f32, layout [M,N] = d_out  (final projection)
// Block: 256 threads = 8 waves; waves tiled 4(M) x 2(N); wave tile 32x32.
// Block tile 128x64; K staged through double-buffered LDS via async copies.
// Grid: (4096/128, 1024/64) = (32, 16).
// ---------------------------------------------------------------------------
__global__ __launch_bounds__(256) void gemm_wmma(const _Float16* __restrict__ A,
                                                 const _Float16* __restrict__ W,
                                                 const float* __restrict__ bias,
                                                 void* __restrict__ out,
                                                 int mode) {
    __shared__ __align__(16) _Float16 Asm[2][BLK_M * BLK_K];   // 2 x 8 KB
    __shared__ __align__(16) _Float16 Bsm[2][BLK_N * BLK_K];   // 2 x 4 KB

    const int K = HDIM, N = HDIM;
    int tid  = threadIdx.x;
    int wave = tid >> 5;
    int lane = tid & 31;
    int lcol = lane & 15;
    int khalf = lane >> 4;
    int kOff = khalf * 8;

    int mBlk = blockIdx.x * BLK_M;
    int nBlk = blockIdx.y * BLK_N;
    int mW = (wave & 3) * 32;      // wave-local M offset
    int nW = (wave >> 2) * 32;     // wave-local N offset

    // Async staging map: 16B chunks; each tile row is BLK_K halves = 64B = 4 chunks.
    // A tile: 128 rows * 4 chunks = 512 -> 2 chunks/thread. B tile: 256 -> 1/thread.
    int arow0 = tid >> 2;                 // 0..63
    int arow1 = arow0 + 64;               // 64..127
    int acol  = (tid & 3) * 8;            // half-element offset within row
    int brow  = tid >> 2;                 // 0..63
    int bcol  = acol;

    auto issue_tiles = [&](int buf, int k0) {
        async_copy_b128(lds_addr(&Asm[buf][arow0 * BLK_K + acol]),
                        A + (size_t)(mBlk + arow0) * K + k0 + acol);
        async_copy_b128(lds_addr(&Asm[buf][arow1 * BLK_K + acol]),
                        A + (size_t)(mBlk + arow1) * K + k0 + acol);
        async_copy_b128(lds_addr(&Bsm[buf][brow * BLK_K + bcol]),
                        W + (size_t)(nBlk + brow) * K + k0 + bcol);
    };

    v8f acc[2][2] = {};

    issue_tiles(0, 0);
    for (int k0 = 0; k0 < K; k0 += BLK_K) {
        int buf = (k0 / BLK_K) & 1;
        wait_async0();
        __syncthreads();
        if (k0 + BLK_K < K) issue_tiles(buf ^ 1, k0 + BLK_K);

        const _Float16* As = Asm[buf];
        const _Float16* Bs = Bsm[buf];
        v16h a0 = load_frag(As + (mW + lcol) * BLK_K + kOff);
        v16h a1 = load_frag(As + (mW + 16 + lcol) * BLK_K + kOff);
        v16h b0 = load_frag(Bs + (nW + lcol) * BLK_K + kOff);
        v16h b1 = load_frag(Bs + (nW + 16 + lcol) * BLK_K + kOff);
        acc[0][0] = wmma_f16(a0, b0, acc[0][0]);
        acc[0][1] = wmma_f16(a0, b1, acc[0][1]);
        acc[1][0] = wmma_f16(a1, b0, acc[1][0]);
        acc[1][1] = wmma_f16(a1, b1, acc[1][1]);
        __syncthreads();
    }

#pragma unroll
    for (int i = 0; i < 2; ++i) {
#pragma unroll
        for (int j = 0; j < 2; ++j) {
            int col = nBlk + nW + j * 16 + lcol;
            float bc = bias[col];
#pragma unroll
            for (int r = 0; r < 8; ++r) {
                int row = mBlk + mW + i * 16 + r + khalf * 8;  // C layout: M = r (+8 hi half)
                float v = acc[i][j][r] + bc;
                if (mode == 2) {
                    ((float*)out)[(size_t)row * N + col] = v;
                } else {
                    int b = row >> 11, s = row & 2047;
                    int h = col >> 6,  d = col & 63;
                    size_t dst = (mode == 0)
                        ? ((size_t)((b * HEADS + h) * S_LEN + s) * DEPTH + d)
                        : ((size_t)((b * HEADS + h) * DEPTH + d) * S_LEN + s);
                    ((_Float16*)out)[dst] = (_Float16)v;
                }
            }
        }
    }
}

// ---------------------------------------------------------------------------
// Flash attention: one wave per (bh, 16-query block).
//   Q,K: [B,H,S,D] f16.  Vt: [B,H,D,S] f16.  O: [B,S,HD] f16.
// Online softmax, f32 stats, WMMA for QK^T and PV.
// Grid: (S/16, B*H) = (128, 32); block = 32 threads (1 wave).
// ---------------------------------------------------------------------------
__global__ __launch_bounds__(32) void attn_wmma(const _Float16* __restrict__ Q,
                                                const _Float16* __restrict__ Kt,
                                                const _Float16* __restrict__ Vt,
                                                _Float16* __restrict__ O) {
    __shared__ __align__(16) _Float16 Pld[16 * 32];

    int bh = blockIdx.y;          // b*16 + h
    int qb = blockIdx.x;          // query block
    int lane = threadIdx.x & 31;
    int lcol = lane & 15;
    int khalf = lane >> 4;
    int kOff = khalf * 8;

    const _Float16* Qb = Q  + (size_t)(bh * S_LEN + qb * 16) * DEPTH;
    const _Float16* Kb = Kt + (size_t)bh * S_LEN * DEPTH;
    const _Float16* Vb = Vt + (size_t)bh * DEPTH * S_LEN;

    // Q A-fragments for this 16-row block (K-dim = 64 -> two 32-chunks), reused.
    v16h aq0 = load_frag(Qb + (size_t)lcol * DEPTH + 0  + kOff);
    v16h aq1 = load_frag(Qb + (size_t)lcol * DEPTH + 32 + kOff);

    float mrow[8], lrow[8];
#pragma unroll
    for (int r = 0; r < 8; ++r) { mrow[r] = -1e30f; lrow[r] = 0.0f; }
    v8f acc[4] = {};              // 16 x 64 output, four 16x16 tiles

    const float scale = 0.125f;   // 1/sqrt(64)

    for (int k0 = 0; k0 < S_LEN; k0 += 32) {
        // ---- logits: 16 queries x 32 keys, two 16x16 WMMA C tiles ----
        v8f s0 = {}, s1 = {};
        {
            const _Float16* kp0 = Kb + (size_t)(k0 + lcol) * DEPTH;
            v16h b0 = load_frag(kp0 + 0 + kOff);
            v16h b1 = load_frag(kp0 + 32 + kOff);
            s0 = wmma_f16(aq0, b0, s0);
            s0 = wmma_f16(aq1, b1, s0);
        }
        {
            const _Float16* kp1 = Kb + (size_t)(k0 + 16 + lcol) * DEPTH;
            v16h b0 = load_frag(kp1 + 0 + kOff);
            v16h b1 = load_frag(kp1 + 32 + kOff);
            s1 = wmma_f16(aq0, b0, s1);
            s1 = wmma_f16(aq1, b1, s1);
        }

        // ---- online softmax update (per-row; rows live in 16-lane halves) ----
#pragma unroll
        for (int r = 0; r < 8; ++r) {
            float e0 = s0[r] * scale;
            float e1 = s1[r] * scale;
            float t = fmaxf(e0, e1);
#pragma unroll
            for (int msk = 1; msk < 16; msk <<= 1)
                t = fmaxf(t, __shfl_xor(t, msk, 32));
            float mnew = fmaxf(mrow[r], t);
            float corr = __expf(mrow[r] - mnew);
            mrow[r] = mnew;

            float p0 = __expf(e0 - mnew);
            float p1 = __expf(e1 - mnew);
            int prow = r + khalf * 8;
            Pld[prow * 32 + lcol]      = (_Float16)p0;
            Pld[prow * 32 + 16 + lcol] = (_Float16)p1;

            float ls = p0 + p1;
#pragma unroll
            for (int msk = 1; msk < 16; msk <<= 1)
                ls += __shfl_xor(ls, msk, 32);
            lrow[r] = lrow[r] * corr + ls;

#pragma unroll
            for (int dt = 0; dt < 4; ++dt)
                acc[dt][r] *= corr;
        }
        __syncthreads();

        // ---- PV: P (16x32) as A-fragment from LDS, V from transposed layout ----
        v16h ap = load_frag(&Pld[lcol * 32 + kOff]);
#pragma unroll
        for (int dt = 0; dt < 4; ++dt) {
            v16h bv = load_frag(Vb + (size_t)(dt * 16 + lcol) * S_LEN + k0 + kOff);
            acc[dt] = wmma_f16(ap, bv, acc[dt]);
        }
        __syncthreads();
    }

    // ---- epilogue: normalize and write [B,S,HD] f16 ----
    int b = bh >> 4, h = bh & 15;
#pragma unroll
    for (int dt = 0; dt < 4; ++dt) {
#pragma unroll
        for (int r = 0; r < 8; ++r) {
            int s = qb * 16 + r + khalf * 8;
            int d = dt * 16 + lcol;
            float v = acc[dt][r] / lrow[r];
            O[(size_t)(b * S_LEN + s) * HDIM + h * DEPTH + d] = (_Float16)v;
        }
    }
}

// ---------------------------------------------------------------------------
// Host launcher
// ---------------------------------------------------------------------------
extern "C" void kernel_launch(void* const* d_in, const int* in_sizes, int n_in,
                              void* d_out, int out_size, void* d_ws, size_t ws_size,
                              hipStream_t stream) {
    const float* x    = (const float*)d_in[0];
    const float* wq_w = (const float*)d_in[1];
    const float* wq_b = (const float*)d_in[2];
    const float* wk_w = (const float*)d_in[3];
    const float* wk_b = (const float*)d_in[4];
    const float* wv_w = (const float*)d_in[5];
    const float* wv_b = (const float*)d_in[6];
    const float* fc_w = (const float*)d_in[7];
    const float* fc_b = (const float*)d_in[8];

    char* ws = (char*)d_ws;
    const size_t MB = 1024 * 1024;
    _Float16* xh   = (_Float16*)(ws + 0);        //  8 MB: x f16 [4096,1024]
    _Float16* wqh  = (_Float16*)(ws + 8  * MB);  //  2 MB
    _Float16* wkh  = (_Float16*)(ws + 10 * MB);  //  2 MB
    _Float16* wvh  = (_Float16*)(ws + 12 * MB);  //  2 MB
    _Float16* fch  = (_Float16*)(ws + 14 * MB);  //  2 MB
    _Float16* Qb   = (_Float16*)(ws + 16 * MB);  //  8 MB: [B,H,S,D]
    _Float16* Kb   = (_Float16*)(ws + 24 * MB);  //  8 MB: [B,H,S,D]
    _Float16* Vtb  = (_Float16*)(ws + 32 * MB);  //  8 MB: [B,H,D,S]
    _Float16* Ab   = (_Float16*)(ws + 40 * MB);  //  8 MB: attn out [B,S,HD]

    // f32 -> f16 conversions
    cvt_f32_f16<<<(MROWS * HDIM + 255) / 256, 256, 0, stream>>>(x, xh, MROWS * HDIM);
    cvt_f32_f16<<<(HDIM * HDIM + 255) / 256, 256, 0, stream>>>(wq_w, wqh, HDIM * HDIM);
    cvt_f32_f16<<<(HDIM * HDIM + 255) / 256, 256, 0, stream>>>(wk_w, wkh, HDIM * HDIM);
    cvt_f32_f16<<<(HDIM * HDIM + 255) / 256, 256, 0, stream>>>(wv_w, wvh, HDIM * HDIM);
    cvt_f32_f16<<<(HDIM * HDIM + 255) / 256, 256, 0, stream>>>(fc_w, fch, HDIM * HDIM);

    dim3 ggrid(MROWS / BLK_M, HDIM / BLK_N);
    // Q/K: [B,H,S,D]; V: transposed [B,H,D,S]
    gemm_wmma<<<ggrid, 256, 0, stream>>>(xh, wqh, wq_b, (void*)Qb, 0);
    gemm_wmma<<<ggrid, 256, 0, stream>>>(xh, wkh, wk_b, (void*)Kb, 0);
    gemm_wmma<<<ggrid, 256, 0, stream>>>(xh, wvh, wv_b, (void*)Vtb, 1);

    attn_wmma<<<dim3(S_LEN / 16, 2 * HEADS), 32, 0, stream>>>(Qb, Kb, Vtb, Ab);

    // Final projection -> f32 d_out
    gemm_wmma<<<ggrid, 256, 0, stream>>>(Ab, fch, fc_b, d_out, 2);
}
